// SourceWindowEncoder_19808389169803
// MI455X (gfx1250) — matrix operand order repaired and checked by reading, not compiled, not run on hardware
//
#include <hip/hip_runtime.h>
#include <hip/hip_bf16.h>

typedef __bf16 bf16;
typedef __attribute__((ext_vector_type(8)))  bf16  bf16x8;
typedef __attribute__((ext_vector_type(16))) bf16  bf16x16;
typedef __attribute__((ext_vector_type(8)))  float f32x8;

#define S_LEN   512
#define BATCH   8
#define NROWS   4096      // S * B
#define HID     512
#define G4      2048      // 4 * HID
#define LWIN    17
#define WINR    8
#define MBLK    128
#define JBLK    32
#define LDSTR   40        // padded LDS row stride (80B: 16B-aligned, conflict-free)
#define PADR    64        // pad rows on each side of G0 for the shifted-view trick

// ---------------- device helpers ----------------

__device__ __forceinline__ float sigm(float x) { return 1.0f / (1.0f + __expf(-x)); }

__device__ __forceinline__ bf16x16 combine16(bf16x8 lo, bf16x8 hi) {
  bf16x16 r;
#pragma unroll
  for (int i = 0; i < 8; ++i) { r[i] = lo[i]; r[i + 8] = hi[i]; }
  return r;
}

// A fragment (16x32 MxK) from LDS tile; CDNA5 layout:
// lanes 0-15: M=lane, K-chunks {0..7, 16..23}; lanes 16-31: M=lane-16, K {8..15, 24..31}
__device__ __forceinline__ bf16x16 load_a_frag_lds(const bf16* lds, int row0, int lane) {
  const int r  = row0 + (lane & 15);
  const int kh = (lane >> 4) * 8;
  const bf16* p = lds + r * LDSTR;
  return combine16(*(const bf16x8*)(p + kh), *(const bf16x8*)(p + 16 + kh));
}

// B fragment (32x16 KxN) from row-major weights W[n][k] (B[k][n] = W[n][k]):
// lane n (0-15) holds K {0..7, 16..23}, lane n+16 holds K {8..15, 24..31}
__device__ __forceinline__ bf16x16 load_b_frag(const bf16* __restrict__ W, int wstride,
                                               int n0, int k0, int lane) {
  const bf16* p = W + (size_t)(n0 + (lane & 15)) * wstride + (k0 + (lane >> 4) * 8);
  return combine16(*(const bf16x8*)p, *(const bf16x8*)(p + 16));
}

// One K=512 GEMM phase: acc[mt][g] += A[128x512] * W[g*512+j][512]^T  (bf16 -> f32)
__device__ __forceinline__ void gemm_phase_k512(
    const bf16* __restrict__ A, int mbase,
    const bf16* __restrict__ W, int wstride,
    int jbase, f32x8 acc[2][4], bf16* lds)
{
  const int tid  = threadIdx.x;
  const int lane = tid & 31;
  const int wave = tid >> 5;
  const int wm = wave & 3;
  const int wj = wave >> 2;
  for (int k0 = 0; k0 < HID; k0 += 32) {
    __syncthreads();                       // protect LDS from previous iteration readers
#pragma unroll
    for (int i = 0; i < 2; ++i) {          // stage A[mbase..+127][k0..k0+31] into LDS
      int idx = tid + i * 256;             // 0..511
      int r = idx >> 2;                    // 0..127
      int q = (idx & 3) * 8;               // 0,8,16,24
      *(bf16x8*)(lds + r * LDSTR + q) =
          *(const bf16x8*)(A + (size_t)(mbase + r) * HID + k0 + q);
    }
    __syncthreads();
    bf16x16 bfrag[4];
#pragma unroll
    for (int g = 0; g < 4; ++g)
      bfrag[g] = load_b_frag(W, wstride, g * HID + jbase + wj * 16, k0, lane);
#pragma unroll
    for (int mt = 0; mt < 2; ++mt) {
      bf16x16 afrag = load_a_frag_lds(lds, wm * 32 + mt * 16, lane);
#pragma unroll
      for (int g = 0; g < 4; ++g)
        acc[mt][g] = __builtin_amdgcn_wmma_f32_16x16x32_bf16(
            false, afrag, false, bfrag[g], (short)0, acc[mt][g], false, false);
    }
  }
}

__device__ __forceinline__ void zero_acc(f32x8 acc[2][4]) {
  const f32x8 z = {0.f, 0.f, 0.f, 0.f, 0.f, 0.f, 0.f, 0.f};
#pragma unroll
  for (int i = 0; i < 2; ++i)
#pragma unroll
    for (int g = 0; g < 4; ++g) acc[i][g] = z;
}

// ---------------- kernels ----------------

// emb[m][e] = bf16(table[ids[m]][e]);  m = s*8+b matches flat [S,B] order
__global__ __launch_bounds__(256) void embed_kernel(
    const int* __restrict__ ids, const float* __restrict__ table, bf16* __restrict__ emb)
{
  const int m  = blockIdx.x;
  const int id = ids[m];
  for (int e = threadIdx.x; e < HID; e += 256)
    emb[(size_t)m * HID + e] = (bf16)table[(size_t)id * HID + e];
}

__global__ __launch_bounds__(256) void f32_to_bf16_kernel(
    const float* __restrict__ src, bf16* __restrict__ dst, int n)
{
  int i = blockIdx.x * 256 + threadIdx.x;
  if (i < n) dst[i] = (bf16)src[i];
}

__global__ __launch_bounds__(256) void zero_bf16_kernel(bf16* __restrict__ dst, int n)
{
  int i = blockIdx.x * 256 + threadIdx.x;
  if (i < n) dst[i] = (bf16)0.0f;
}

// G0 = emb @ wih0^T + bias : [4096 x 2048] bf16 (col = gate*512 + j)
__global__ __launch_bounds__(256) void g0_gemm_kernel(
    const bf16* __restrict__ emb, const bf16* __restrict__ wih,
    const float* __restrict__ bias, bf16* __restrict__ G0)
{
  __shared__ bf16 lds[MBLK * LDSTR];
  f32x8 acc[2][4];
  zero_acc(acc);
  const int mbase = blockIdx.x * MBLK;
  const int jbase = blockIdx.y * JBLK;
  gemm_phase_k512(emb, mbase, wih, HID, jbase, acc, lds);

  const int lane = threadIdx.x & 31;
  const int wave = threadIdx.x >> 5;
  const int wm = wave & 3, wj = wave >> 2;
  const int jcol = jbase + wj * 16 + (lane & 15);
#pragma unroll
  for (int mt = 0; mt < 2; ++mt) {
    const int m0 = mbase + wm * 32 + mt * 16 + ((lane >> 4) << 3);   // D-layout M base
#pragma unroll
    for (int g = 0; g < 4; ++g) {
      const float bv = bias[g * HID + jcol];
#pragma unroll
      for (int r = 0; r < 8; ++r)
        G0[(size_t)(m0 + r) * G4 + g * HID + jcol] = (bf16)(acc[mt][g][r] + bv);
    }
  }
}

// Fused LSTM step: gates = sum_phases(A_p @ W_p^T) + addend_rows/bias ; cell ; mask-carry
__global__ __launch_bounds__(256) void lstm_step_kernel(
    const bf16* __restrict__ A0, const bf16* __restrict__ W0, int ws0,
    const bf16* __restrict__ A1, const bf16* __restrict__ W1, int ws1,
    const bf16* __restrict__ A2, const bf16* __restrict__ W2, int ws2,
    int nphases,
    const bf16*  __restrict__ addend_rows,   // [rows][2048], pre-offset view (layer 0) or null
    const float* __restrict__ bias,          // [2048] (layer 1) or null
    const bf16*  __restrict__ h_prev,        // [4096][512] carry source, or null (=0)
    const float* __restrict__ c_prev,        // [4096][512], or null (=0)
    bf16*  __restrict__ h_out,               // [4096][512]
    float* __restrict__ c_out,               // [4096][512]
    float* __restrict__ h_out_f32,           // optional f32 copy of h, or null
    int lo, int hi)                          // row valid iff lo <= s < hi  (s = m>>3)
{
  __shared__ bf16 lds[MBLK * LDSTR];
  f32x8 acc[2][4];
  zero_acc(acc);
  const int mbase = blockIdx.x * MBLK;
  const int jbase = blockIdx.y * JBLK;
  if (nphases > 0) gemm_phase_k512(A0, mbase, W0, ws0, jbase, acc, lds);
  if (nphases > 1) gemm_phase_k512(A1, mbase, W1, ws1, jbase, acc, lds);
  if (nphases > 2) gemm_phase_k512(A2, mbase, W2, ws2, jbase, acc, lds);

  const int lane = threadIdx.x & 31;
  const int wave = threadIdx.x >> 5;
  const int wm = wave & 3, wj = wave >> 2;
  const int jcol = jbase + wj * 16 + (lane & 15);

  float bi = 0.f, bff = 0.f, bg = 0.f, bo = 0.f;
  if (bias) {
    bi = bias[jcol];           bff = bias[HID + jcol];
    bg = bias[2 * HID + jcol]; bo  = bias[3 * HID + jcol];
  }

#pragma unroll
  for (int mt = 0; mt < 2; ++mt) {
    const int m0 = mbase + wm * 32 + mt * 16 + ((lane >> 4) << 3);
#pragma unroll
    for (int r = 0; r < 8; ++r) {
      const int m = m0 + r;
      float gi = acc[mt][0][r] + bi;
      float gf = acc[mt][1][r] + bff;
      float gg = acc[mt][2][r] + bg;
      float go = acc[mt][3][r] + bo;
      if (addend_rows) {                       // layer-0: precomputed input gates (shifted view)
        const bf16* ar = addend_rows + (size_t)m * G4;
        gi += (float)ar[jcol];
        gf += (float)ar[HID + jcol];
        gg += (float)ar[2 * HID + jcol];
        go += (float)ar[3 * HID + jcol];
      }
      const size_t off = (size_t)m * HID + jcol;
      const float cp = c_prev ? c_prev[off] : 0.0f;
      float cn = sigm(gf) * cp + sigm(gi) * tanhf(gg);
      float hn = sigm(go) * tanhf(cn);
      const int s = m >> 3;
      if (s < lo || s >= hi) {                 // masked step: carry state unchanged
        cn = cp;
        hn = h_prev ? (float)h_prev[off] : 0.0f;
      }
      c_out[off] = cn;
      h_out[off] = (bf16)hn;
      if (h_out_f32) h_out_f32[off] = hn;
    }
  }
}

__global__ __launch_bounds__(256) void final_avg_kernel(
    const float* __restrict__ hf, const float* __restrict__ hb,
    float* __restrict__ out, int n)
{
  int i = blockIdx.x * 256 + threadIdx.x;
  if (i < n) out[i] = 0.5f * (hf[i] + hb[i]);
}

// ---------------- host ----------------

static inline int imax_(int a, int b) { return a > b ? a : b; }
static inline int imin_(int a, int b) { return a < b ? a : b; }

extern "C" void kernel_launch(void* const* d_in, const int* in_sizes, int n_in,
                              void* d_out, int out_size, void* d_ws, size_t ws_size,
                              hipStream_t stream) {
  (void)in_sizes; (void)n_in; (void)out_size; (void)ws_size;
  const int*   ids   = (const int*)d_in[0];
  const float* table = (const float*)d_in[1];
  const float* wih_f[2][2]  = {{(const float*)d_in[2],  (const float*)d_in[5]},
                               {(const float*)d_in[8],  (const float*)d_in[11]}};
  const float* whh_f[2][2]  = {{(const float*)d_in[3],  (const float*)d_in[6]},
                               {(const float*)d_in[9],  (const float*)d_in[12]}};
  const float* bias_f[2][2] = {{(const float*)d_in[4],  (const float*)d_in[7]},
                               {(const float*)d_in[10], (const float*)d_in[13]}};

  char* p = (char*)d_ws;
  auto carve = [&](size_t bytes) -> char* {
    char* r = p; p += (bytes + 255) & ~(size_t)255; return r;
  };
  bf16* emb = (bf16*)carve((size_t)NROWS * HID * sizeof(bf16));
  bf16* G0pad[2];
  for (int d = 0; d < 2; ++d)
    G0pad[d] = (bf16*)carve((size_t)(NROWS + 2 * PADR) * G4 * sizeof(bf16));
  bf16 *wihb[2][2], *whhb[2][2];
  for (int l = 0; l < 2; ++l)
    for (int d = 0; d < 2; ++d) {
      const int indim = (l == 0) ? HID : 2 * HID;
      wihb[l][d] = (bf16*)carve((size_t)G4 * indim * sizeof(bf16));
      whhb[l][d] = (bf16*)carve((size_t)G4 * HID * sizeof(bf16));
    }
  bf16* Y0[2];
  for (int d = 0; d < 2; ++d)
    Y0[d] = (bf16*)carve((size_t)LWIN * NROWS * HID * sizeof(bf16));
  float* cbuf[2];
  for (int i = 0; i < 2; ++i) cbuf[i] = (float*)carve((size_t)NROWS * HID * sizeof(float));
  bf16* h1pp[2];
  for (int i = 0; i < 2; ++i) h1pp[i] = (bf16*)carve((size_t)NROWS * HID * sizeof(bf16));
  float* h1f32[2];
  for (int d = 0; d < 2; ++d) h1f32[d] = (float*)carve((size_t)NROWS * HID * sizeof(float));

  // 1. embedding gather -> bf16
  embed_kernel<<<NROWS, 256, 0, stream>>>(ids, table, emb);

  // 2. weights f32 -> bf16
  for (int l = 0; l < 2; ++l)
    for (int d = 0; d < 2; ++d) {
      const int indim = (l == 0) ? HID : 2 * HID;
      const int n1 = G4 * indim;
      f32_to_bf16_kernel<<<(n1 + 255) / 256, 256, 0, stream>>>(wih_f[l][d], wihb[l][d], n1);
      const int n2 = G4 * HID;
      f32_to_bf16_kernel<<<(n2 + 255) / 256, 256, 0, stream>>>(whh_f[l][d], whhb[l][d], n2);
    }

  // 3. G0 = emb @ wih0^T + b0 (once per token: 17x cheaper than per-slot), padded view
  for (int d = 0; d < 2; ++d) {
    const int np = PADR * G4;
    zero_bf16_kernel<<<(np + 255) / 256, 256, 0, stream>>>(G0pad[d], np);
    zero_bf16_kernel<<<(np + 255) / 256, 256, 0, stream>>>(
        G0pad[d] + (size_t)(PADR + NROWS) * G4, np);
    g0_gemm_kernel<<<dim3(32, 16), 256, 0, stream>>>(
        emb, wihb[0][d], bias_f[0][d], G0pad[d] + (size_t)PADR * G4);
  }

  // 4. layer-0 recurrences (17 steps each direction); ys[t] stored in Y0[d][t]
  for (int d = 0; d < 2; ++d) {
    for (int step = 0; step < LWIN; ++step) {
      const int t     = (d == 0) ? step : (LWIN - 1 - step);
      const int tprev = (d == 0) ? (t - 1) : (t + 1);
      const bf16*  hin = (step == 0) ? nullptr : Y0[d] + (size_t)tprev * NROWS * HID;
      const float* cin = (step == 0) ? nullptr : cbuf[(step + 1) & 1];
      bf16* hout = Y0[d] + (size_t)t * NROWS * HID;
      const int rowoff = PADR + (t - WINR) * BATCH;            // shifted-view gather
      const bf16* addend = G0pad[d] + (size_t)rowoff * G4;
      const int lo = imax_(0, WINR - t);
      const int hi = imin_(S_LEN, S_LEN + WINR - t);
      lstm_step_kernel<<<dim3(32, 16), 256, 0, stream>>>(
          hin, whhb[0][d], HID,
          nullptr, nullptr, 0,
          nullptr, nullptr, 0,
          (step == 0) ? 0 : 1,
          addend, nullptr,
          hin, cin, hout, cbuf[step & 1], nullptr, lo, hi);
    }
  }

  // 5. layer-1 recurrences: only slots up to the center (9 steps each direction)
  for (int d = 0; d < 2; ++d) {
    for (int step = 0; step <= WINR; ++step) {
      const int t = (d == 0) ? step : (LWIN - 1 - step);
      const bf16*  hin = (step == 0) ? nullptr : h1pp[(step + 1) & 1];
      const float* cin = (step == 0) ? nullptr : cbuf[(step + 1) & 1];
      const int lo = imax_(0, WINR - t);
      const int hi = imin_(S_LEN, S_LEN + WINR - t);
      lstm_step_kernel<<<dim3(32, 16), 256, 0, stream>>>(
          Y0[0] + (size_t)t * NROWS * HID, wihb[1][d],       2 * HID,   // yf * wih1[:, :512]
          Y0[1] + (size_t)t * NROWS * HID, wihb[1][d] + HID, 2 * HID,   // yb * wih1[:, 512:]
          hin,                              whhb[1][d],      HID,       // h  * whh1
          (step == 0) ? 2 : 3,
          nullptr, bias_f[1][d],
          hin, cin, h1pp[step & 1], cbuf[step & 1], h1f32[d], lo, hi);
    }
  }

  // 6. out = (yf1[WIN] + yb1[WIN]) / 2  (f32 copies hold the final/center-slot h)
  final_avg_kernel<<<(NROWS * HID) / 256, 256, 0, stream>>>(
      h1f32[0], h1f32[1], (float*)d_out, NROWS * HID);
}
